// MessagePassingLayer_32976758898973
// MI455X (gfx1250) — matrix operand order repaired and checked by reading, not compiled, not run on hardware
//
#include <hip/hip_runtime.h>
#include <hip/hip_bf16.h>
#include <stdint.h>

#define N_NODES   50000
#define N_EDGES   250000
#define DIM       256
#define KTOT      768          // 3 fused K-blocks of 256 (Hh, E, Hh*E)
#define EDGE_TILE 80           // 250000 % 80 == 0 -> no tail handling
#define MT        5            // M-tiles (of 16 edges) per wave
#define NT        2            // N-tiles (of 16 cols) per wave
#define ETHREADS  256          // 8 waves (wave32) x NT n-tiles = 16 n-tiles

typedef __attribute__((ext_vector_type(16))) __bf16 v16bf;
typedef __attribute__((ext_vector_type(8)))  float  v8f;

struct __align__(16) U128 { unsigned int x, y, z, w; };
struct __align__(32) B256 { U128 lo, hi; };

__device__ __forceinline__ unsigned short f2bf(float f) {
  unsigned int u = __float_as_uint(f);
  u += 0x7FFFu + ((u >> 16) & 1u);       // round-to-nearest-even
  return (unsigned short)(u >> 16);
}

// Packed f32x2 -> bf16x2 (lowers to v_cvt_pk_bf16_f32 on gfx1250).
__device__ __forceinline__ unsigned int bfpack(float a, float b) {
  __hip_bfloat162 h = __float22bfloat162_rn(float2{a, b});
  unsigned int u;
  __builtin_memcpy(&u, &h, sizeof(u));   // __hip_bfloat162 not trivially copyable
  return u;
}

// Generic pointer to LDS: low 32 bits are the workgroup-relative LDS offset.
__device__ __forceinline__ unsigned int lds_off(const void* p) {
  return (unsigned int)(unsigned long long)p;
}

// ---------------------------------------------------------------- zero scratch
__global__ void gnn_zero_kernel(float4* __restrict__ p, int n4) {
  int i = blockIdx.x * blockDim.x + threadIdx.x;
  int stride = gridDim.x * blockDim.x;
  float4 z = make_float4(0.f, 0.f, 0.f, 0.f);
  for (; i < n4; i += stride) p[i] = z;
}

// ------------------------------------------------- fold weights -> bf16 B-tiles
// WcT[dir][n][k] (row stride KTOT): k<512 -> W[n][k] + W[n][512+(k&255)]
// (Hh and E blocks with the shared (Hh+E) block folded in);
// k in [512,768) -> W[n][256+k] (the Hh*E block).
__global__ void gnn_prep_w_kernel(const float* __restrict__ W_fwd,
                                  const float* __restrict__ W_back,
                                  unsigned short* __restrict__ WcT) {
  int idx = blockIdx.x * blockDim.x + threadIdx.x;
  const int per_dir = DIM * KTOT;
  if (idx >= 2 * per_dir) return;
  int dir = idx / per_dir;
  int rem = idx - dir * per_dir;
  int n = rem / KTOT;
  int k = rem - n * KTOT;
  const float* W = dir ? W_back : W_fwd;
  float v;
  if (k < 512) v = W[n * 1024 + k] + W[n * 1024 + 512 + (k & 255)];
  else         v = W[n * 1024 + 256 + k];
  WcT[idx] = f2bf(v);
}

// ------------------------------------------------------------- edge GEMM+scatter
__launch_bounds__(ETHREADS)
__global__ void gnn_edge_kernel(const float* __restrict__ H,
                                const float* __restrict__ E,
                                const int*   __restrict__ ht,
                                const unsigned short* __restrict__ WcT,
                                const float* __restrict__ b_fwd,
                                const float* __restrict__ b_back,
                                float* __restrict__ agg,
                                float* __restrict__ cnt) {
  const int dir = blockIdx.y;               // 0: fwd (src=head), 1: back (src=tail)
  const int e0  = blockIdx.x * EDGE_TILE;
  const int t   = threadIdx.x;

  // bf16 operand tiles: [0]=Hh, [1]=E, [2]=Hh*E  (3 x 80 x 264 x 2B ~ 127 KB)
  __shared__ unsigned short sA[3][EDGE_TILE][DIM + 8];
  __shared__ int s_ht[EDGE_TILE * 2];       // (head,tail) pairs, async-filled

  // Async-copy the 80x2 int32 ht tile into LDS (ASYNCcnt-tracked CDNA5 path).
  if (t < (EDGE_TILE * 2) / 4) {            // 40 x b128
    unsigned long long ga = (unsigned long long)(ht + 2 * e0 + t * 4);
    unsigned int       la = lds_off(&s_ht[0]) + t * 16;
    asm volatile("global_load_async_to_lds_b128 %0, %1, off"
                 :: "v"(la), "v"(ga) : "memory");
  }
  // Prefetch next tile of E rows (gfx1250 global_prefetch path).
  if (e0 + EDGE_TILE < N_EDGES)
    __builtin_prefetch(E + (size_t)(e0 + EDGE_TILE) * DIM + t * 80, 0, 0);

  asm volatile("s_wait_asynccnt 0x0" ::: "memory");
  __syncthreads();

  // Build bf16 operand tiles: 640 chunks of (row, 32 cols), 256 threads.
  for (int chunk = t; chunk < EDGE_TILE * 8; chunk += ETHREADS) {
    const int m  = chunk >> 3;
    const int c0 = (chunk & 7) * 32;
    const int src = s_ht[2 * m + dir];
    const float* hrow = H + (size_t)src * DIM + c0;
    const float* erow = E + (size_t)(e0 + m) * DIM + c0;
    unsigned int* ph = (unsigned int*)&sA[0][m][c0];
    unsigned int* pe = (unsigned int*)&sA[1][m][c0];
    unsigned int* px = (unsigned int*)&sA[2][m][c0];
#pragma unroll
    for (int j = 0; j < 32; j += 8) {
      float4 h0 = *(const float4*)(hrow + j);
      float4 h1 = *(const float4*)(hrow + j + 4);
      float4 g0 = *(const float4*)(erow + j);
      float4 g1 = *(const float4*)(erow + j + 4);
      float hv[8] = {h0.x, h0.y, h0.z, h0.w, h1.x, h1.y, h1.z, h1.w};
      float ev[8] = {g0.x, g0.y, g0.z, g0.w, g1.x, g1.y, g1.z, g1.w};
#pragma unroll
      for (int i = 0; i < 8; i += 2) {
        int q = (j + i) >> 1;
        ph[q] = bfpack(hv[i], hv[i + 1]);
        pe[q] = bfpack(ev[i], ev[i + 1]);
        px[q] = bfpack(hv[i] * ev[i], hv[i + 1] * ev[i + 1]);
      }
    }
  }
  __syncthreads();

  // 8 waves; each owns NT=2 16-column N-tiles and MT=5 M-tiles; K = 768.
  // Each A fragment feeds NT WMMAs from registers -> LDS traffic per WMMA halves.
  const int wave = t >> 5;
  const int lane = t & 31;
  const int n0   = wave * (16 * NT);
  const int mrow = lane & 15;     // A row / B column within tile (ISA layouts)
  const int hl   = lane >> 4;     // half-wave selects K window

  v8f acc[NT][MT];
#pragma unroll
  for (int nt = 0; nt < NT; ++nt)
#pragma unroll
    for (int mt = 0; mt < MT; ++mt) acc[nt][mt] = (v8f){};

  const unsigned short* Wb = WcT + (size_t)dir * DIM * KTOT
                                 + (size_t)(n0 + mrow) * KTOT + hl * 16;
#pragma unroll
  for (int term = 0; term < 3; ++term) {
#pragma unroll
    for (int kk = 0; kk < DIM; kk += 32) {
      // NT B fragments (16 contiguous K bf16 per lane), reused for MT WMMAs each.
      v16bf bv[NT];
#pragma unroll
      for (int nt = 0; nt < NT; ++nt) {
        const unsigned short* bp = Wb + (size_t)(nt * 16) * KTOT + term * DIM + kk;
        B256 fb;
        fb.lo = *(const U128*)(bp);
        fb.hi = *(const U128*)(bp + 8);
        bv[nt] = __builtin_bit_cast(v16bf, fb);
      }
      // Hoist all A fragments into distinct registers so the backend issues
      // 10 ds_load_b128 back-to-back and drains dscnt incrementally.
      v16bf av[MT];
#pragma unroll
      for (int mt = 0; mt < MT; ++mt) {
        const unsigned short* ap = &sA[term][mt * 16 + mrow][kk + hl * 16];
        B256 fa;
        fa.lo = *(const U128*)(ap);        // K = kk + hl*16     .. +7
        fa.hi = *(const U128*)(ap + 8);    // K = kk + hl*16 + 8 .. +7
        av[mt] = __builtin_bit_cast(v16bf, fa);
      }
#pragma unroll
      for (int mt = 0; mt < MT; ++mt)
#pragma unroll
        for (int nt = 0; nt < NT; ++nt)
          acc[nt][mt] = __builtin_amdgcn_wmma_f32_16x16x32_bf16(
              false, av[mt], false, bv[nt], (short)0, acc[nt][mt], false, false);
    }
  }

  // Epilogue: bias + scatter-add into L2-resident accumulator.
#pragma unroll
  for (int nt = 0; nt < NT; ++nt) {
    const int ncol = n0 + nt * 16 + mrow;
    const float bias = (dir == 0 ? b_fwd : b_back)[ncol];
#pragma unroll
    for (int mt = 0; mt < MT; ++mt) {
#pragma unroll
      for (int r = 0; r < 8; ++r) {
        int m = mt * 16 + hl * 8 + r;      // C/D layout: VGPR r, half -> M
        int dst = s_ht[2 * m + (1 - dir)];
        atomicAdd(&agg[(size_t)dst * DIM + ncol], acc[nt][mt][r] + bias);
      }
    }
  }
  if (t < EDGE_TILE) atomicAdd(&cnt[s_ht[2 * t + (1 - dir)]], 1.0f);
}

// ------------------------------------------------------ mean/leaky/LN epilogue
__global__ void gnn_finalize_kernel(const float* __restrict__ H,
                                    const float* __restrict__ agg,
                                    const float* __restrict__ cnt,
                                    const float* __restrict__ gamma,
                                    const float* __restrict__ beta,
                                    float* __restrict__ out) {
  const int node = blockIdx.x;
  const int c    = threadIdx.x;
  const size_t idx = (size_t)node * DIM + c;

  float denom = fmaxf(cnt[node], 1.0f);
  float x = agg[idx] / denom;
  x = (x > 0.f) ? x : 0.01f * x;           // leaky_relu
  x += H[idx];

  __shared__ float rsum[8], rsum2[8];
  float s = x, s2 = x * x;
#pragma unroll
  for (int o = 16; o > 0; o >>= 1) {
    s  += __shfl_xor(s,  o, 32);
    s2 += __shfl_xor(s2, o, 32);
  }
  if ((c & 31) == 0) { rsum[c >> 5] = s; rsum2[c >> 5] = s2; }
  __syncthreads();

  __shared__ float s_mu, s_rstd;
  if (c == 0) {
    float t1 = 0.f, t2 = 0.f;
#pragma unroll
    for (int i = 0; i < 8; ++i) { t1 += rsum[i]; t2 += rsum2[i]; }
    float mu  = t1 * (1.0f / DIM);
    float var = t2 * (1.0f / DIM) - mu * mu;
    s_mu = mu;
    s_rstd = rsqrtf(var + 1e-5f);
  }
  __syncthreads();

  out[idx] = (x - s_mu) * s_rstd * gamma[c] + beta[c];
}

// ---------------------------------------------------------------------- launch
extern "C" void kernel_launch(void* const* d_in, const int* in_sizes, int n_in,
                              void* d_out, int out_size, void* d_ws, size_t ws_size,
                              hipStream_t stream) {
  const float* H      = (const float*)d_in[0];
  const float* E      = (const float*)d_in[1];
  const int*   ht     = (const int*)d_in[2];
  const float* W_fwd  = (const float*)d_in[3];
  const float* b_fwd  = (const float*)d_in[4];
  const float* W_back = (const float*)d_in[5];
  const float* b_back = (const float*)d_in[6];
  const float* gamma  = (const float*)d_in[7];
  const float* beta   = (const float*)d_in[8];
  float* out = (float*)d_out;

  // Workspace layout: agg[50000*256] f32 | cnt[50000] f32 | WcT[2*256*768] bf16
  float* agg = (float*)d_ws;
  float* cnt = agg + (size_t)N_NODES * DIM;
  unsigned short* WcT = (unsigned short*)(cnt + N_NODES);

  const int n4 = (N_NODES * DIM + N_NODES) / 4;   // zero agg+cnt together
  hipLaunchKernelGGL(gnn_zero_kernel, dim3(2048), dim3(256), 0, stream,
                     (float4*)d_ws, n4);
  hipLaunchKernelGGL(gnn_prep_w_kernel, dim3((2 * DIM * KTOT + 255) / 256),
                     dim3(256), 0, stream, W_fwd, W_back, WcT);
  hipLaunchKernelGGL(gnn_edge_kernel, dim3(N_EDGES / EDGE_TILE, 2),
                     dim3(ETHREADS), 0, stream,
                     H, E, ht, WcT, b_fwd, b_back, agg, cnt);
  hipLaunchKernelGGL(gnn_finalize_kernel, dim3(N_NODES), dim3(DIM), 0, stream,
                     H, agg, cnt, gamma, beta, out);
}
